// FasterRcnnPredictor_22101901705390
// MI455X (gfx1250) — compile-verified, matrix operand order checked by reading, not run on hardware
//
#include <hip/hip_runtime.h>
#include <hip/hip_bf16.h>
#include <stdint.h>

typedef __attribute__((ext_vector_type(16))) _Float16 v16h;
typedef __attribute__((ext_vector_type(8)))  float    v8f;
typedef __attribute__((ext_vector_type(4)))  int      v4i;

typedef __attribute__((address_space(1))) v4i as1_v4i;
typedef __attribute__((address_space(3))) v4i as3_v4i;

#define NBOX      8000
#define NPAD      8192
#define NEG_INF_F (-1e30f)
#define KEEP_THR  (-5e29f)

#if __has_builtin(__builtin_amdgcn_global_load_async_to_lds_b128)
#define HAVE_ASYNC_LDS 1
#else
#define HAVE_ASYNC_LDS 0
#endif

__device__ __forceinline__ void wait_asynccnt0() {
#if __has_builtin(__builtin_amdgcn_s_wait_asynccnt)
  __builtin_amdgcn_s_wait_asynccnt(0);
#else
  asm volatile("s_wait_asynccnt 0" ::: "memory");
#endif
}

// ---------------------------------------------------------------------------
// K1: masked-score argsort (descending, stable) via bitonic sort of packed
//     u64 keys ((sortable_score)<<32 | idx) wholly inside LDS (64KB of 320KB).
// ---------------------------------------------------------------------------
__global__ __launch_bounds__(1024) void nms_sort_kernel(
    const float* __restrict__ scores, int* __restrict__ order) {
  extern __shared__ unsigned long long key[];   // NPAD * 8 = 64 KB dynamic LDS
  const int tid = threadIdx.x;

  for (int i = tid; i < NPAD; i += 1024) {
    float s = NEG_INF_F;
    if (i < NBOX) {
      float v = scores[i];
      if (v >= 0.5f) s = v;
    }
    uint32_t b   = __float_as_uint(s);
    uint32_t f2u = b ^ ((b & 0x80000000u) ? 0xFFFFFFFFu : 0x80000000u); // ascending map
    uint32_t k32 = ~f2u;                                               // descending
    key[i] = ((unsigned long long)k32 << 32) | (uint32_t)i;            // stable tiebreak
  }

  for (int k = 2; k <= NPAD; k <<= 1) {
    for (int j = k >> 1; j > 0; j >>= 1) {
      __syncthreads();
      for (int i = tid; i < NPAD; i += 1024) {
        int ixj = i ^ j;
        if (ixj > i) {
          unsigned long long a = key[i], b2 = key[ixj];
          bool up = ((i & k) == 0);
          if ((a > b2) == up) { key[i] = b2; key[ixj] = a; }
        }
      }
    }
  }
  __syncthreads();
  for (int i = tid; i < NPAD; i += 1024)
    order[i] = (int)(key[i] & 0xFFFFFFFFu);
}

// ---------------------------------------------------------------------------
// K2: gather boxes/labels/masked-scores into sorted order (+padding rows).
// ---------------------------------------------------------------------------
__global__ __launch_bounds__(256) void nms_gather_kernel(
    const float* __restrict__ boxes, const float* __restrict__ scores,
    const int* __restrict__ labels, const int* __restrict__ order,
    float4* __restrict__ bsort, int* __restrict__ lsort,
    float* __restrict__ scsort) {
  int r = blockIdx.x * 256 + threadIdx.x;
  if (r >= NPAD) return;
  int o = order[r];
  float4 bx = {0.f, 0.f, 0.f, 0.f};
  int lab = -1;
  float s = NEG_INF_F;
  if (o < NBOX) {
    bx  = ((const float4*)boxes)[o];
    lab = labels[o];
    float v = scores[o];
    if (v >= 0.5f) s = v;
  }
  bsort[r]  = bx;
  lsort[r]  = lab;
  scsort[r] = s;
}

// ---------------------------------------------------------------------------
// K3: suppression bit-matrix. One wave per 16x16 tile.
//     Label equality = onehot(l_i) @ onehot(l_j)^T via v_wmma_f32_16x16x32_f16
//     (co-executes with the VALU IoU math). Since NUM_CLASSES=6 < 8, only
//     K=0..7 can ever be hot: lanes 0-15 build 8 half-slots, lanes 16-31 are
//     all-zero (sentinel label) -> 8 cmp+sel per matrix instead of 32.
//     The 16 i-tile boxes are staged global->LDS with the gfx1250 async-LDS
//     path (ASYNCcnt) when the builtin is available.
//     suppress[i][j] = same_label && IoU>=0.5 && j>i   (u16 words, 512/row)
// ---------------------------------------------------------------------------
__global__ __launch_bounds__(256) void nms_supp_kernel(
    const float4* __restrict__ bsort, const int* __restrict__ lsort,
    unsigned short* __restrict__ supp) {
  const int tid    = threadIdx.x;
  const int lane   = tid & 31;
  const int wave   = tid >> 5;
  const int tile_i = blockIdx.x >> 6;                 // 512 tile rows
  const int jblk   = blockIdx.x & 63;                 // 64 blocks of 8 tiles
  const int i0     = tile_i * 16;
  const int j0     = (jblk * 8 + wave) * 16;

  __shared__ float4 bi_s[16];
#if HAVE_ASYNC_LDS
  if (tid < 16) {
    __builtin_amdgcn_global_load_async_to_lds_b128(
        (as1_v4i*)(bsort + i0 + tid), (as3_v4i*)(&bi_s[tid]),
        /*offset=*/0, /*cpol=*/0);
  }
  wait_asynccnt0();
  __syncthreads();
#else
  if (tid < 16) bi_s[tid] = bsort[i0 + tid];
  __syncthreads();
#endif

  unsigned short result = 0;

  if (j0 + 15 > i0) {   // tiles entirely at/below diagonal are all-zero
    const int  m       = lane & 15;
    const bool lowHalf = (lane < 16);
    // Sentinel (never matches h in 0..7) zeroes the upper-lane slots, which
    // hold K>=8 and can never equal a label in [0,6).
    const int rowSel = lowHalf ? lsort[i0 + m] : 0x7fff;
    const int colSel = lowHalf ? lsort[j0 + m] : 0x7fff;

    v16h a = {};   // slots 8..15 (K=16..23 / 24..31) always zero
    v16h b = {};
#pragma unroll
    for (int h = 0; h < 8; ++h) {     // lanes 0-15: slot h -> K=h
      a[h] = (h == rowSel) ? (_Float16)1.0f : (_Float16)0.0f;
      b[h] = (h == colSel) ? (_Float16)1.0f : (_Float16)0.0f;
    }
    v8f c = {};
    c = __builtin_amdgcn_wmma_f32_16x16x32_f16(
        /*neg_a=*/false, a, /*neg_b=*/false, b,
        /*c_mod=*/(short)0, c, /*reuse_a=*/false, /*reuse_b=*/false);

    // This lane's column box (N = lane&15)
    const float4 bj = bsort[j0 + m];
    const float  aj = fmaxf(bj.z - bj.x, 0.f) * fmaxf(bj.w - bj.y, 0.f);
    const int    j  = j0 + m;
    const int    kbase = lowHalf ? 0 : 8;

    uint32_t bal[8];
#pragma unroll
    for (int r = 0; r < 8; ++r) {
      const int    M  = r + kbase;     // rows r (lanes 0-15) / r+8 (16-31)
      const float4 bi = bi_s[M];
      float ai   = fmaxf(bi.z - bi.x, 0.f) * fmaxf(bi.w - bi.y, 0.f);
      float ix1  = fmaxf(bi.x, bj.x);
      float iy1  = fmaxf(bi.y, bj.y);
      float ix2  = fminf(bi.z, bj.z);
      float iy2  = fminf(bi.w, bj.w);
      float inter = fmaxf(ix2 - ix1, 0.f) * fmaxf(iy2 - iy1, 0.f);
      float iou   = inter / (ai + aj - inter + 1e-6f);
      bool  sup   = (c[r] > 0.5f) && (iou >= 0.5f) && (j > (i0 + M));
      bal[r] = __builtin_amdgcn_ballot_w32(sup);
    }

    // Ballots are wave-uniform: lane t picks the word for output row i0+t.
    // row t<8 -> bal[t] low half; row t>=8 -> bal[t-8] high half.
    uint32_t w = bal[0];
    const int s = lane & 7;
    w = (s == 1) ? bal[1] : w;
    w = (s == 2) ? bal[2] : w;
    w = (s == 3) ? bal[3] : w;
    w = (s == 4) ? bal[4] : w;
    w = (s == 5) ? bal[5] : w;
    w = (s == 6) ? bal[6] : w;
    w = (s == 7) ? bal[7] : w;
    result = (lane & 8) ? (unsigned short)(w >> 16)
                        : (unsigned short)(w & 0xFFFFu);
  }

  if (lane < 16)
    supp[(size_t)(i0 + lane) * 512 + (j0 >> 4)] = result;
}

// ---------------------------------------------------------------------------
// K4: serial greedy suppression scan. 8192-bit removed mask lives in LDS
//     (256 u32, one per thread). Next row prefetched to hide L2 latency.
// ---------------------------------------------------------------------------
__global__ __launch_bounds__(256) void nms_scan_kernel(
    const float* __restrict__ scsort, const uint32_t* __restrict__ supp32,
    uint32_t* __restrict__ removed_out) {
  __shared__ uint32_t removed[256];
  __shared__ int flag;
  const int tid = threadIdx.x;
  removed[tid] = 0;
  __syncthreads();

  for (int i = 0; i < NBOX; ++i) {
    if (i + 1 < NBOX)
      __builtin_prefetch(&supp32[(size_t)(i + 1) * 256 + tid], 0, 3);
    if (tid == 0) {
      bool k0 = scsort[i] > KEEP_THR;
      flag = k0 && !((removed[i >> 5] >> (i & 31)) & 1u);
    }
    __syncthreads();
    if (flag)
      removed[tid] |= supp32[(size_t)i * 256 + tid];
    __syncthreads();
  }
  removed_out[tid] = removed[tid];
}

// ---------------------------------------------------------------------------
// K5: masked outputs, concatenated as float:
//     [0,4N) boxes | [4N,5N) labels | [5N,6N) scores | [6N,7N) keep
// ---------------------------------------------------------------------------
__global__ __launch_bounds__(256) void nms_out_kernel(
    const float4* __restrict__ bsort, const int* __restrict__ lsort,
    const float* __restrict__ scsort, const uint32_t* __restrict__ removed,
    float* __restrict__ out) {
  int r = blockIdx.x * 256 + threadIdx.x;
  if (r >= NBOX) return;
  float s    = scsort[r];
  bool  keep = (s > KEEP_THR) && !((removed[r >> 5] >> (r & 31)) & 1u);
  float4 bx  = bsort[r];
  out[r * 4 + 0] = keep ? bx.x : 0.f;
  out[r * 4 + 1] = keep ? bx.y : 0.f;
  out[r * 4 + 2] = keep ? bx.z : 0.f;
  out[r * 4 + 3] = keep ? bx.w : 0.f;
  out[4 * NBOX + r] = keep ? (float)lsort[r] : -1.0f;
  out[5 * NBOX + r] = keep ? s : 0.f;
  out[6 * NBOX + r] = keep ? 1.f : 0.f;
}

// ---------------------------------------------------------------------------
extern "C" void kernel_launch(void* const* d_in, const int* in_sizes, int n_in,
                              void* d_out, int out_size, void* d_ws, size_t ws_size,
                              hipStream_t stream) {
  const float* boxes  = (const float*)d_in[0];   // N x 4
  const float* scores = (const float*)d_in[1];   // N
  const int*   labels = (const int*)d_in[2];     // N
  float*       out    = (float*)d_out;

  char* ws = (char*)d_ws;
  int*            order   = (int*)(ws + 0);                 //  32 KB
  float4*         bsort   = (float4*)(ws + 32768);          // 128 KB
  int*            lsort   = (int*)(ws + 163840);            //  32 KB
  float*          scsort  = (float*)(ws + 196608);          //  32 KB
  uint32_t*       removed = (uint32_t*)(ws + 229376);       //   1 KB
  unsigned short* supp    = (unsigned short*)(ws + 262144); //   8 MB

  nms_sort_kernel<<<1, 1024, NPAD * sizeof(unsigned long long), stream>>>(
      scores, order);
  nms_gather_kernel<<<NPAD / 256, 256, 0, stream>>>(
      boxes, scores, labels, order, bsort, lsort, scsort);
  nms_supp_kernel<<<512 * 64, 256, 0, stream>>>(bsort, lsort, supp);
  nms_scan_kernel<<<1, 256, 0, stream>>>(scsort, (const uint32_t*)supp, removed);
  nms_out_kernel<<<(NBOX + 255) / 256, 256, 0, stream>>>(
      bsort, lsort, scsort, removed, out);
}
